// GAT_584115553007
// MI455X (gfx1250) — matrix operand order repaired and checked by reading, not compiled, not run on hardware
//
#include <hip/hip_runtime.h>
#include <hip/hip_bf16.h>

typedef __attribute__((ext_vector_type(16))) _Float16 v16h;
typedef __attribute__((ext_vector_type(8)))  float    v8f;

#define NNODES 50000
#define NEDGES 1600000
#define ETOT   (NEDGES + NNODES)
#define FDIM   128          // IN = H*HID = 128 for both layers' K
#define HEADS1 8
#define HID1   16
#define OUTC   16
#define NEG_SLOPE 0.2f

// ---------------------------------------------------------------------------
// WMMA GEMM: Out[n, o] = sum_k X[n,k] * W[o,k]   (K fixed = 128)
// RT row-tiles x CT col-tiles per block; 8 waves (256 threads), one wave per
// (row-tile, col-tile). X tile staged to LDS as f16; W gathered (L2-resident).
// ---------------------------------------------------------------------------
template <int RT, int CT, int OutF>
__global__ __launch_bounds__(256) void gat_wmma_gemm(
    const float* __restrict__ X, const float* __restrict__ W,
    float* __restrict__ Out, int nrows)
{
    __shared__ _Float16 tile[RT * 16 * 128];

    const int wave = threadIdx.x >> 5;
    const int lane = threadIdx.x & 31;
    const int rowblk = blockIdx.x * (RT * 16);

    // Stage X tile (f32 -> f16), zero-pad past nrows.
    const int total = RT * 16 * 128;
    for (int i = threadIdx.x; i < total; i += 256) {
        const int r = i >> 7, c = i & 127;
        const int gr = rowblk + r;
        float v = (gr < nrows) ? X[(size_t)gr * 128 + c] : 0.0f;
        tile[i] = (_Float16)v;
    }
    __syncthreads();

    const int rt = wave / CT;
    const int ct = wave % CT;
    const int colbase = ct * 16;

    // Per-lane fragment coordinates (ISA 7.12.2, wave32, 16-bit A / B):
    const int arow   = rt * 16 + (lane & 15);     // A: M = lane&15
    const int akbase = (lane >> 4) * 8;           // A: lanes 16..31 start at K=8
    const int bcol   = lane & 15;                 // B: N = lane&15
    const int bkbase = (lane >> 4) * 8;           // B: lanes 16..31 start at K=8

    v8f acc = {};
#pragma unroll
    for (int kt = 0; kt < 4; ++kt) {
        const int kb = kt * 32;
        v16h a, b;
#pragma unroll
        for (int j = 0; j < 16; ++j) {
            const int ka = kb + akbase + j + ((j >= 8) ? 8 : 0);
            a[j] = tile[arow * 128 + ka];
            const int kbb = kb + bkbase + j + ((j >= 8) ? 8 : 0);
            b[j] = (_Float16)W[(colbase + bcol) * 128 + kbb];
        }
        acc = __builtin_amdgcn_wmma_f32_16x16x32_f16(
            /*neg_a=*/false, a, /*neg_b=*/false, b,
            /*c_mod=*/(short)0, acc, /*reuse_a=*/false, /*reuse_b=*/false);
    }

    // C/D layout: reg r, lane L -> M = r + 8*(L>>4), N = L&15
    const int mofs = (lane >> 4) * 8;
    const int row0 = rowblk + rt * 16;
#pragma unroll
    for (int r = 0; r < 8; ++r) {
        const int gr = row0 + mofs + r;
        if (gr < nrows)
            Out[(size_t)gr * OutF + colbase + bcol] = acc[r];
    }
}

// ---------------------------------------------------------------------------
// Per-(node,head) attention coefficients: as[i]=dot(h_i,att_src), ad likewise.
// h layout [N, Hh*C]; i = node*Hh + head  ->  h + i*C is the head slice.
// ---------------------------------------------------------------------------
template <int Hh, int C>
__global__ void gat_node_att(const float* __restrict__ h,
                             const float* __restrict__ att_s,
                             const float* __restrict__ att_d,
                             float* __restrict__ as_, float* __restrict__ ad_,
                             int nitems)
{
    int i = blockIdx.x * blockDim.x + threadIdx.x;
    if (i >= nitems) return;
    const int head = i & (Hh - 1);                // Hh is a power of two
    const float* hv = h + (size_t)i * C;
    float s = 0.0f, d = 0.0f;
#pragma unroll
    for (int c = 0; c < C; ++c) {
        const float v = hv[c];
        s += v * att_s[head * C + c];
        d += v * att_d[head * C + c];
    }
    as_[i] = s;
    ad_[i] = d;
}

__device__ __forceinline__ float gat_leaky(float a) {
    return (a > 0.0f) ? a : a * NEG_SLOPE;
}

// Pass 1: softmax denominators per (dst,head). Shift-free softmax (logits <<1).
template <int Hh>
__global__ void gat_edge_denom(const int* __restrict__ src,
                               const int* __restrict__ dst,
                               const float* __restrict__ as_,
                               const float* __restrict__ ad_,
                               float* __restrict__ denom)
{
    int t = blockIdx.x * blockDim.x + threadIdx.x;
    if (t >= ETOT * Hh) return;
    const int e = t / Hh;                         // Hh = 1 or 8 -> shift
    const int hh = t & (Hh - 1);
    const int s = (e < NEDGES) ? src[e] : (e - NEDGES);
    const int d = (e < NEDGES) ? dst[e] : (e - NEDGES);
    const float a = gat_leaky(as_[d * Hh + hh] + ad_[s * Hh + hh]);
    atomicAdd(&denom[d * Hh + hh], __expf(a));
}

// Pass 2: weighted scatter-add aggregation: agg[dst] += alpha * h[src].
template <int Hh, int C>
__global__ void gat_edge_aggr(const int* __restrict__ src,
                              const int* __restrict__ dst,
                              const float* __restrict__ as_,
                              const float* __restrict__ ad_,
                              const float* __restrict__ denom,
                              const float* __restrict__ h,
                              float* __restrict__ agg)
{
    int t = blockIdx.x * blockDim.x + threadIdx.x;
    if (t >= ETOT * Hh) return;
    const int e = t / Hh;
    const int hh = t & (Hh - 1);
    const int s = (e < NEDGES) ? src[e] : (e - NEDGES);
    const int d = (e < NEDGES) ? dst[e] : (e - NEDGES);
    const float a = gat_leaky(as_[d * Hh + hh] + ad_[s * Hh + hh]);
    const float alpha = __expf(a) / denom[d * Hh + hh];
    const float* hv = h + ((size_t)s * Hh + hh) * C;
    float* ov = agg + ((size_t)d * Hh + hh) * C;
#pragma unroll
    for (int c = 0; c < C; ++c)
        atomicAdd(&ov[c], hv[c] * alpha);
}

// Layer-1 epilogue: out = elu(agg + bias)   (F = 128, mask instead of mod)
__global__ void gat_bias_elu(const float* __restrict__ agg,
                             const float* __restrict__ b,
                             float* __restrict__ out, int total)
{
    int i = blockIdx.x * blockDim.x + threadIdx.x;
    if (i >= total) return;
    const float v = agg[i] + b[i & (FDIM - 1)];
    out[i] = (v > 0.0f) ? v : (__expf(v) - 1.0f);
}

// Layer-2 epilogue: log_softmax(agg + bias) over 16 classes.
__global__ void gat_final_lsm(const float* __restrict__ agg,
                              const float* __restrict__ b,
                              float* __restrict__ out, int n)
{
    int i = blockIdx.x * blockDim.x + threadIdx.x;
    if (i >= n) return;
    float v[OUTC];
    float m = -1e30f;
#pragma unroll
    for (int c = 0; c < OUTC; ++c) {
        v[c] = agg[(size_t)i * OUTC + c] + b[c];
        m = fmaxf(m, v[c]);
    }
    float sum = 0.0f;
#pragma unroll
    for (int c = 0; c < OUTC; ++c) sum += __expf(v[c] - m);
    const float lse = m + __logf(sum);
#pragma unroll
    for (int c = 0; c < OUTC; ++c)
        out[(size_t)i * OUTC + c] = v[c] - lse;
}

// ---------------------------------------------------------------------------
extern "C" void kernel_launch(void* const* d_in, const int* in_sizes, int n_in,
                              void* d_out, int out_size, void* d_ws, size_t ws_size,
                              hipStream_t stream)
{
    const float* x      = (const float*)d_in[0];
    const int*   ei     = (const int*)  d_in[1];
    const float* W1     = (const float*)d_in[2];
    const float* att_s1 = (const float*)d_in[3];
    const float* att_d1 = (const float*)d_in[4];
    const float* b1     = (const float*)d_in[5];
    const float* W2     = (const float*)d_in[6];
    const float* att_s2 = (const float*)d_in[7];
    const float* att_d2 = (const float*)d_in[8];
    const float* b2     = (const float*)d_in[9];

    const int* esrc = ei;
    const int* edst = ei + NEDGES;

    // Workspace carve-up (256B aligned)
    char* ws = (char*)d_ws;
    size_t off = 0;
    auto carve = [&](size_t bytes) -> char* {
        char* p = ws + off;
        off = (off + bytes + 255) & ~(size_t)255;
        return p;
    };
    float* h1   = (float*)carve((size_t)NNODES * FDIM * 4);  // 25.6 MB (later: ELU out)
    float* as1  = (float*)carve((size_t)NNODES * HEADS1 * 4);
    float* ad1  = (float*)carve((size_t)NNODES * HEADS1 * 4);
    float* den1 = (float*)carve((size_t)NNODES * HEADS1 * 4);
    float* big  = (float*)carve((size_t)NNODES * FDIM * 4);  // agg1; reused below
    float* agg1 = big;
    float* g2   = big;                                        // reuse after ELU
    float* agg2 = big + (size_t)NNODES * OUTC;
    float* as2  = as1;                                        // N*4 fits in N*8*4
    float* ad2  = ad1;
    float* den2 = den1;

    auto cdiv = [](long long a, long long b) { return (int)((a + b - 1) / b); };

    // ---- Layer 1 ----
    // h1 = x @ W1^T   [50000 x 128], one block = 16 rows, 8 col-tiles
    gat_wmma_gemm<1, 8, FDIM><<<cdiv(NNODES, 16), 256, 0, stream>>>(x, W1, h1, NNODES);

    gat_node_att<HEADS1, HID1><<<cdiv((long long)NNODES * HEADS1, 256), 256, 0, stream>>>(
        h1, att_s1, att_d1, as1, ad1, NNODES * HEADS1);

    hipMemsetAsync(den1, 0, (size_t)NNODES * HEADS1 * 4, stream);
    gat_edge_denom<HEADS1><<<cdiv((long long)ETOT * HEADS1, 256), 256, 0, stream>>>(
        esrc, edst, as1, ad1, den1);

    hipMemsetAsync(agg1, 0, (size_t)NNODES * FDIM * 4, stream);
    gat_edge_aggr<HEADS1, HID1><<<cdiv((long long)ETOT * HEADS1, 256), 256, 0, stream>>>(
        esrc, edst, as1, ad1, den1, h1, agg1);

    // ELU(agg1 + b1) -> overwrite h1 buffer (h1 no longer needed)
    gat_bias_elu<<<cdiv((long long)NNODES * FDIM, 256), 256, 0, stream>>>(
        agg1, b1, h1, NNODES * FDIM);

    // ---- Layer 2 ----
    // g2 = h @ W2^T   [50000 x 16], one block = 128 rows, 1 col-tile
    gat_wmma_gemm<8, 1, OUTC><<<cdiv(NNODES, 128), 256, 0, stream>>>(h1, W2, g2, NNODES);

    gat_node_att<1, OUTC><<<cdiv(NNODES, 256), 256, 0, stream>>>(
        g2, att_s2, att_d2, as2, ad2, NNODES);

    hipMemsetAsync(den2, 0, (size_t)NNODES * 4, stream);
    gat_edge_denom<1><<<cdiv(ETOT, 256), 256, 0, stream>>>(esrc, edst, as2, ad2, den2);

    hipMemsetAsync(agg2, 0, (size_t)NNODES * OUTC * 4, stream);
    gat_edge_aggr<1, OUTC><<<cdiv(ETOT, 256), 256, 0, stream>>>(
        esrc, edst, as2, ad2, den2, g2, agg2);

    gat_final_lsm<<<cdiv(NNODES, 256), 256, 0, stream>>>(agg2, b2, (float*)d_out, NNODES);
}
